// RIKPConvFPN_56023553409406
// MI455X (gfx1250) — matrix operand order repaired and checked by reading, not compiled, not run on hardware
//
#include <hip/hip_runtime.h>
#include <hip/hip_bf16.h>
#include <math.h>

typedef __attribute__((ext_vector_type(16))) _Float16 v16h;
typedef __attribute__((ext_vector_type(8)))  _Float16 v8h;
typedef __attribute__((ext_vector_type(4)))  _Float16 v4h;
typedef __attribute__((ext_vector_type(8)))  float    v8f;

#define CN0 40000
#define CN1 10000
#define CN2 2500
#define KNB 32
#define NKP 15
#define SIGMA 0.05f
#define GROUPS 32
#define NEGV -1000000000.0f
#define EPSV 1e-5f
#define SLOPE 0.1f

__device__ __forceinline__ float act_fn(float x, int act) {
  return (act == 1) ? (x > 0.f ? x : SLOPE * x) : x;
}

// ---------------------------------------------------------------------------
// Kernel-point generation: numpy RandomState(42) MT19937 + legacy rk_gauss
// ---------------------------------------------------------------------------
struct MTState { unsigned mt[624]; int mti; int has_g; double g; };

__device__ unsigned mt_next(MTState& s) {
  if (s.mti >= 624) {
    for (int i = 0; i < 624; i++) {
      unsigned y = (s.mt[i] & 0x80000000u) | (s.mt[(i + 1) % 624] & 0x7fffffffu);
      s.mt[i] = s.mt[(i + 397) % 624] ^ (y >> 1) ^ ((y & 1u) ? 0x9908b0dfu : 0u);
    }
    s.mti = 0;
  }
  unsigned y = s.mt[s.mti++];
  y ^= y >> 11; y ^= (y << 7) & 0x9d2c5680u; y ^= (y << 15) & 0xefc60000u; y ^= y >> 18;
  return y;
}
__device__ double mt_double(MTState& s) {
  unsigned a = mt_next(s) >> 5, b = mt_next(s) >> 6;
  return (a * 67108864.0 + b) / 9007199254740992.0;
}
__device__ double mt_gauss(MTState& s) {
  if (s.has_g) { s.has_g = 0; return s.g; }
  double x1, x2, r2;
  do {
    x1 = 2.0 * mt_double(s) - 1.0;
    x2 = 2.0 * mt_double(s) - 1.0;
    r2 = x1 * x1 + x2 * x2;
  } while (r2 >= 1.0 || r2 == 0.0);
  double f = sqrt(-2.0 * log(r2) / r2);
  s.g = f * x1; s.has_g = 1;
  return f * x2;
}

// kp[0..44] = kp(R), kp[45..89] = kp(2R), kp[90..134] = kp(4R)
__global__ void kp_gen_kernel(float* kp) {
  if (threadIdx.x != 0 || blockIdx.x != 0) return;
  MTState s;
  s.mt[0] = 42u;
  for (int i = 1; i < 624; i++)
    s.mt[i] = 1812433253u * (s.mt[i - 1] ^ (s.mt[i - 1] >> 30)) + (unsigned)i;
  s.mti = 624; s.has_g = 0; s.g = 0.0;
  double u[NKP][3];
  for (int i = 0; i < NKP; i++)
    for (int j = 0; j < 3; j++) u[i][j] = mt_gauss(s);
  for (int i = 0; i < NKP; i++) {
    double nn = sqrt(u[i][0]*u[i][0] + u[i][1]*u[i][1] + u[i][2]*u[i][2]) + 1e-9;
    for (int j = 0; j < 3; j++) u[i][j] /= nn;
  }
  double r[NKP];
  for (int i = 0; i < NKP; i++) r[i] = cbrt(mt_double(s));
  for (int i = 0; i < NKP; i++)
    for (int j = 0; j < 3; j++) {
      float base = (float)(u[i][j] * r[i]);
      kp[i*3 + j]        = 0.0625f * base;
      kp[45 + i*3 + j]   = 0.125f  * base;
      kp[90 + i*3 + j]   = 0.25f   * base;
    }
}

// ---------------------------------------------------------------------------
// Small utility kernels
// ---------------------------------------------------------------------------
__global__ void zero_kernel(float* p, int n) {
  int i = threadIdx.x; if (i < n) p[i] = 0.f;
}

__global__ void mean_reduce_kernel(const float* __restrict__ pts, int n, float* acc) {
  __shared__ float sx[256], sy[256], sz[256];
  int t = threadIdx.x;
  float ax = 0.f, ay = 0.f, az = 0.f;
  for (int i = blockIdx.x * 256 + t; i < n; i += gridDim.x * 256) {
    ax += pts[i*3]; ay += pts[i*3 + 1]; az += pts[i*3 + 2];
  }
  sx[t] = ax; sy[t] = ay; sz[t] = az; __syncthreads();
  for (int s = 128; s > 0; s >>= 1) {
    if (t < s) { sx[t] += sx[t+s]; sy[t] += sy[t+s]; sz[t] += sz[t+s]; }
    __syncthreads();
  }
  if (t == 0) { atomicAdd(&acc[0], sx[0]); atomicAdd(&acc[1], sy[0]); atomicAdd(&acc[2], sz[0]); }
}

__global__ void copy_int_kernel(const int* __restrict__ src, int* __restrict__ dst, long long total) {
  long long i = blockIdx.x * 256LL + threadIdx.x;
  if (i < total) dst[i] = src[i];
}

__global__ void add_leaky_kernel(const float* __restrict__ a, const float* __restrict__ b,
                                 float* __restrict__ y, long long total) {
  long long i = blockIdx.x * 256LL + threadIdx.x;
  if (i >= total) return;
  float v = a[i] + b[i];
  y[i] = v > 0.f ? v : SLOPE * v;
}

__global__ void maxpool_kernel(const float* __restrict__ X, const int* __restrict__ idx,
                               long long total, int Ns, int C, float* __restrict__ Y) {
  long long i = blockIdx.x * 256LL + threadIdx.x;
  if (i >= total) return;
  int n = (int)(i / C), c = (int)(i % C);
  float m = NEGV;
  for (int k = 0; k < KNB; k++) {
    int j = idx[(size_t)n * KNB + k];
    float v = (j < Ns) ? X[(size_t)j * C + c] : NEGV;
    m = v > m ? v : m;
  }
  Y[i] = m;
}

__global__ void gather_concat_kernel(const float* __restrict__ A, const int* __restrict__ up,
                                     const float* __restrict__ B, float* __restrict__ Y,
                                     long long total, int Ca, int Cb) {
  long long i = blockIdx.x * 256LL + threadIdx.x;
  if (i >= total) return;
  int cw = Ca + Cb;
  int n = (int)(i / cw), c = (int)(i % cw);
  if (c < Ca) Y[i] = A[(size_t)up[n] * Ca + c];
  else        Y[i] = B[(size_t)n * Cb + (c - Ca)];
}

// ---------------------------------------------------------------------------
// Rotation-invariant features + e0 linear (4->64) + relu + max over K
// blockDim = 64, one block per point
// ---------------------------------------------------------------------------
__global__ void rif_feats0_kernel(const float* __restrict__ pts, const int* __restrict__ nb,
                                  const float* __restrict__ macc, int n0,
                                  const float* __restrict__ W, const float* __restrict__ bias,
                                  float* __restrict__ f0) {
  __shared__ float rif[KNB][4];
  int n = blockIdx.x, t = threadIdx.x;
  float mx = macc[0] / n0, my = macc[1] / n0, mz = macc[2] / n0;
  float qx = pts[n*3], qy = pts[n*3+1], qz = pts[n*3+2];
  float cx = qx - mx, cy = qy - my, cz = qz - mz;
  float inv = 1.f / (sqrtf(cx*cx + cy*cy + cz*cz) + 1e-9f);
  float nx = cx*inv, ny = cy*inv, nz = cz*inv;
  if (t < KNB) {
    int j = nb[(size_t)n * KNB + t];
    float px = pts[j*3], py = pts[j*3+1], pz = pts[j*3+2];
    float dx = px - qx, dy = py - qy, dz = pz - qz;
    float d = sqrtf(dx*dx + dy*dy + dz*dz);
    float iv = 1.f / (d + 1e-9f);
    float vx = dx*iv, vy = dy*iv, vz = dz*iv;
    float ex = px - mx, ey = py - my, ez = pz - mz;
    float ie = 1.f / (sqrtf(ex*ex + ey*ey + ez*ez) + 1e-9f);
    float ax = ex*ie, ay = ey*ie, az = ez*ie;
    rif[t][0] = d;
    rif[t][1] = nx*vx + ny*vy + nz*vz;
    rif[t][2] = ax*vx + ay*vy + az*vz;
    rif[t][3] = nx*ax + ny*ay + nz*az;
  }
  __syncthreads();
  int c = t;  // 64 channels
  float w0 = W[c], w1 = W[64 + c], w2 = W[128 + c], w3 = W[192 + c], bb = bias[c];
  float best = 0.f;
  for (int k = 0; k < KNB; k++) {
    float v = rif[k][0]*w0 + rif[k][1]*w1 + rif[k][2]*w2 + rif[k][3]*w3 + bb;
    v = v > 0.f ? v : 0.f;
    best = v > best ? v : best;
  }
  f0[(size_t)n * 64 + c] = best;
}

// ---------------------------------------------------------------------------
// KPConv aggregation: agg[n, p*C + c] = sum_k relu(1-d(n,k,p)/sigma)*f[idx[n,k]][c]
// Output directly as f16 for the WMMA GEMM against K-weights (15C x C).
// blockDim = C (32/64/128), one block per query point.
// ---------------------------------------------------------------------------
__global__ void kpconv_agg_kernel(const float* __restrict__ q, const float* __restrict__ s,
                                  const int* __restrict__ idx, const float* __restrict__ feats,
                                  const float* __restrict__ kp, float sigma,
                                  int Nq, int Ns, int C, _Float16* __restrict__ agg) {
  __shared__ float ws[KNB][16];
  __shared__ int   is[KNB];
  __shared__ float kps[NKP * 3];
  int n = blockIdx.x, t = threadIdx.x;
  for (int i = t; i < NKP * 3; i += blockDim.x) kps[i] = kp[i];
  __syncthreads();
  float qx = q[n*3], qy = q[n*3+1], qz = q[n*3+2];
  if (t < KNB) {
    int j = idx[(size_t)n * KNB + t];
    is[t] = j;
    float sx, sy, sz;
    if (j < Ns) { sx = s[j*3]; sy = s[j*3+1]; sz = s[j*3+2]; }
    else        { sx = sy = sz = 1000000.0f; }
    float rx = sx - qx, ry = sy - qy, rz = sz - qz;
#pragma unroll
    for (int p = 0; p < NKP; p++) {
      float dx = rx - kps[p*3], dy = ry - kps[p*3+1], dz = rz - kps[p*3+2];
      float d = sqrtf(dx*dx + dy*dy + dz*dz);
      float w = 1.f - d / sigma;
      ws[t][p] = w > 0.f ? w : 0.f;
    }
  }
  __syncthreads();
  int c = t;
  float a[NKP];
#pragma unroll
  for (int p = 0; p < NKP; p++) a[p] = 0.f;
  for (int k = 0; k < KNB; k++) {
    int j = is[k];
    float f = (j < Ns) ? feats[(size_t)j * C + c] : 0.f;
    float wv;
#pragma unroll
    for (int p = 0; p < NKP; p++) { wv = ws[k][p]; a[p] += wv * f; }
  }
  size_t base = (size_t)n * (NKP * C) + c;
#pragma unroll
  for (int p = 0; p < NKP; p++) agg[base + (size_t)p * C] = (_Float16)a[p];
}

// ---------------------------------------------------------------------------
// WMMA GEMM: Y[N,Co] = act(X[N,K] @ W[K,Co] + bias)
// Requirements (all call sites satisfy): K % 32 == 0, Co % 32 == 0.
// 256 threads = 8 waves; block tile 64x64. Each wave owns TWO 16-row m-tiles
// sharing one B fragment -> 2 WMMAs per k-step per wave (6 ds_load_b128 per
// 2 WMMAs, half the barrier cost per FLOP vs a 1-tile wave).
// A kept [row][k]; B stored transposed [col][k] so every lane's fragment is
// contiguous 16B runs in LDS -> ds_load_b128. Global staging is float4/b128
// with chunk-granular guards. Fragment layouts per CDNA5 ISA 7.12.2.
// ---------------------------------------------------------------------------
template <int XH>
__global__ void gemm_wmma_kernel(const float* __restrict__ Xf, const _Float16* __restrict__ Xh,
                                 const float* __restrict__ W, const float* __restrict__ bias,
                                 float* __restrict__ Y, int N, int K, int Co, int act) {
  __shared__ _Float16 As[64][40];   // [row][k], 80B row stride (16B aligned chunks)
  __shared__ _Float16 Bt[64][40];   // [col][k], transposed B
  int tid = threadIdx.x;
  int lane = tid & 31, wave = tid >> 5;
  int l15 = lane & 15, hi = lane >> 4;
  int rb = blockIdx.x * 64, cb = blockIdx.y * 64;
  int mA = (wave >> 2) * 32;        // wave's m-tiles: rows mA..mA+15 and mA+16..mA+31
  int n0 = (wave & 3) * 16;

  v8f acc0 = {}, acc1 = {};
  for (int k0 = 0; k0 < K; k0 += 32) {
    // ---- stage A tile (64 rows x 32 k): two float4 chunks per thread ----
#pragma unroll
    for (int i = 0; i < 2; i++) {
      int qq = tid + i * 256;
      int r = qq >> 3, ak = (qq & 7) * 4;
      int gr = rb + r;
      v4h h4;
      if (gr < N) {
        if (XH) {
          h4 = *reinterpret_cast<const v4h*>(Xh + (size_t)gr * K + k0 + ak);
        } else {
          const float4 f4 = *reinterpret_cast<const float4*>(Xf + (size_t)gr * K + k0 + ak);
          h4[0] = (_Float16)f4.x; h4[1] = (_Float16)f4.y;
          h4[2] = (_Float16)f4.z; h4[3] = (_Float16)f4.w;
        }
      } else {
        h4[0] = h4[1] = h4[2] = h4[3] = (_Float16)0.f;
      }
      *reinterpret_cast<v4h*>(&As[r][ak]) = h4;
    }
    // ---- stage B tile (32 k x 64 cols) transposed ----
#pragma unroll
    for (int i = 0; i < 2; i++) {
      int qq = tid + i * 256;
      int kk = qq >> 4, nc = (qq & 15) * 4;
      int gk = k0 + kk, gc = cb + nc;
      float4 f4 = make_float4(0.f, 0.f, 0.f, 0.f);
      if (gc + 3 < Co)
        f4 = *reinterpret_cast<const float4*>(W + (size_t)gk * Co + gc);
      Bt[nc + 0][kk] = (_Float16)f4.x;
      Bt[nc + 1][kk] = (_Float16)f4.y;
      Bt[nc + 2][kk] = (_Float16)f4.z;
      Bt[nc + 3][kk] = (_Float16)f4.w;
    }
    __syncthreads();
    // ---- fragment loads: 6x ds_load_b128 per lane, 2 WMMAs ----
    v8h blo = *reinterpret_cast<const v8h*>(&Bt[n0 + l15][hi * 8]);
    v8h bhi = *reinterpret_cast<const v8h*>(&Bt[n0 + l15][16 + hi * 8]);
    v16h b = __builtin_shufflevector(blo, bhi, 0,1,2,3,4,5,6,7,8,9,10,11,12,13,14,15);
    v8h a0lo = *reinterpret_cast<const v8h*>(&As[mA + l15][hi * 8]);
    v8h a0hi = *reinterpret_cast<const v8h*>(&As[mA + l15][16 + hi * 8]);
    v8h a1lo = *reinterpret_cast<const v8h*>(&As[mA + 16 + l15][hi * 8]);
    v8h a1hi = *reinterpret_cast<const v8h*>(&As[mA + 16 + l15][16 + hi * 8]);
    v16h a0 = __builtin_shufflevector(a0lo, a0hi, 0,1,2,3,4,5,6,7,8,9,10,11,12,13,14,15);
    v16h a1 = __builtin_shufflevector(a1lo, a1hi, 0,1,2,3,4,5,6,7,8,9,10,11,12,13,14,15);
    acc0 = __builtin_amdgcn_wmma_f32_16x16x32_f16(
        false, a0, false, b, (short)0, acc0, false, false);
    acc1 = __builtin_amdgcn_wmma_f32_16x16x32_f16(
        false, a1, false, b, (short)0, acc1, false, false);
    __syncthreads();
  }
  int col = cb + n0 + l15;
  float bv = (bias && col < Co) ? bias[col] : 0.f;
#pragma unroll
  for (int v = 0; v < 8; v++) {
    int row = rb + mA + hi * 8 + v;                 // C/D f32 16x16 layout
    if (row < N && col < Co)
      Y[(size_t)row * Co + col] = act_fn(acc0[v] + bv, act);
  }
#pragma unroll
  for (int v = 0; v < 8; v++) {
    int row = rb + mA + 16 + hi * 8 + v;
    if (row < N && col < Co)
      Y[(size_t)row * Co + col] = act_fn(acc1[v] + bv, act);
  }
}

// ---------------------------------------------------------------------------
// GroupNorm: stats per group (mean/var over N rows x C/G channels), then apply.
// ---------------------------------------------------------------------------
__global__ void gn_stats_kernel(const float* __restrict__ X, int N, int C, int cg,
                                float* __restrict__ stats) {
  __shared__ double ss[256], s2[256];
  int g = blockIdx.x, t = threadIdx.x;
  double a = 0.0, b = 0.0;
  long long M = (long long)N * cg;
  for (long long i = t; i < M; i += 256) {
    int n = (int)(i / cg);
    int c = g * cg + (int)(i % cg);
    double v = X[(size_t)n * C + c];
    a += v; b += v * v;
  }
  ss[t] = a; s2[t] = b; __syncthreads();
  for (int s = 128; s > 0; s >>= 1) {
    if (t < s) { ss[t] += ss[t+s]; s2[t] += s2[t+s]; }
    __syncthreads();
  }
  if (t == 0) {
    double m = ss[0] / (double)M;
    stats[g*2] = (float)m;
    stats[g*2 + 1] = (float)(s2[0] / (double)M - m * m);
  }
}

__global__ void gn_apply_kernel(const float* __restrict__ X, const float* __restrict__ stats,
                                const float* __restrict__ ga, const float* __restrict__ be,
                                float* __restrict__ Y, long long total, int C, int cg, int act) {
  long long i = blockIdx.x * 256LL + threadIdx.x;
  if (i >= total) return;
  int c = (int)(i % C);
  int g = c / cg;
  float m = stats[2*g], v = stats[2*g + 1];
  float y = (X[i] - m) * rsqrtf(v + EPSV) * ga[c] + be[c];
  Y[i] = act_fn(y, act);
}

// ---------------------------------------------------------------------------
// Host orchestration
// ---------------------------------------------------------------------------
extern "C" void kernel_launch(void* const* d_in, const int* in_sizes, int n_in,
                              void* d_out, int out_size, void* d_ws, size_t ws_size,
                              hipStream_t stream) {
  (void)in_sizes; (void)n_in; (void)out_size; (void)ws_size;
  const float* pts0 = (const float*)d_in[0];
  const float* pts1 = (const float*)d_in[1];
  const float* pts2 = (const float*)d_in[2];
  const int* nb0  = (const int*)d_in[3];
  const int* nb1  = (const int*)d_in[4];
  const int* nb2  = (const int*)d_in[5];
  const int* sub0 = (const int*)d_in[6];
  const int* sub1 = (const int*)d_in[7];
  const int* up0  = (const int*)d_in[8];
  const int* up1  = (const int*)d_in[9];
#define PRM(i) ((const float*)d_in[10 + (i)])

  float* out = (float*)d_out;
  float* l1_out = out;                                   // N0 x 256
  float* f0_out = out + (size_t)CN0 * 256;               // N0 x 64
  float* f3_out = f0_out + (size_t)CN0 * 64;             // N2 x 512
  int*   gidx_out = (int*)(f3_out + (size_t)CN2 * 512);  // N0 x 32 (int bits)

  // workspace carve
  char* wp = (char*)d_ws; size_t off = 0;
  auto carve = [&](size_t bytes) -> void* {
    void* p = wp + off; off = (off + bytes + 255) & ~(size_t)255; return p;
  };
  float* kpb   = (float*)carve(3 * 45 * sizeof(float));
  float* meanb = (float*)carve(64);
  float* stats = (float*)carve(GROUPS * 2 * sizeof(float));
  _Float16* agg = (_Float16*)carve((size_t)CN0 * NKP * 64 * sizeof(_Float16)); // 76.8 MB
  const size_t TSZ = (size_t)CN0 * 128 * sizeof(float);
  float* T1 = (float*)carve(TSZ);
  float* T2 = (float*)carve(TSZ);
  float* T3 = (float*)carve(TSZ);
  float* T4 = (float*)carve(TSZ);
  float* T5 = (float*)carve(TSZ);
  float* T6 = (float*)carve(TSZ);
  float* F1 = (float*)carve(TSZ);                        // feats_s1: N0 x 128
  float* B1 = (float*)carve(TSZ);
  float* B2 = (float*)carve(TSZ);
  float* F2 = (float*)carve((size_t)CN1 * 256 * sizeof(float)); // feats_s2

  auto gemmF = [&](const float* X, const float* Wm, const float* bs, float* Yp,
                   int N, int K, int Co, int act) {
    dim3 g((N + 63) / 64, (Co + 63) / 64);
    gemm_wmma_kernel<0><<<g, 256, 0, stream>>>(X, nullptr, Wm, bs, Yp, N, K, Co, act);
  };
  auto gemmH = [&](const _Float16* Xp, const float* Wm, float* Yp, int N, int K, int Co) {
    dim3 g((N + 63) / 64, (Co + 63) / 64);
    gemm_wmma_kernel<1><<<g, 256, 0, stream>>>(nullptr, Xp, Wm, nullptr, Yp, N, K, Co, 0);
  };
  auto gnorm = [&](const float* X, const float* ga, const float* be, float* Yp,
                   int N, int C, int act) {
    int cg = C / GROUPS;
    gn_stats_kernel<<<GROUPS, 256, 0, stream>>>(X, N, C, cg, stats);
    long long tot = (long long)N * C;
    gn_apply_kernel<<<(unsigned)((tot + 255) / 256), 256, 0, stream>>>(
        X, stats, ga, be, Yp, tot, C, cg, act);
  };
  auto kpconv = [&](const float* qp, const float* sp, const int* idx, const float* feats,
                    int Nq, int Ns, int C, const float* kp, float sig,
                    const float* Kw, float* Yp) {
    kpconv_agg_kernel<<<Nq, C, 0, stream>>>(qp, sp, idx, feats, kp, sig, Nq, Ns, C, agg);
    gemmH(agg, Kw, Yp, Nq, NKP * C, C);
  };
  auto resb = [&](const float* x, int Nin, int Nout,
                  const float* qp, const float* sp, const int* idx,
                  int pu1, int pK, int pu2, int psc, bool strided,
                  int Ci, int mid, int Co, const float* kp, float sig, float* outp) {
    gemmF(x, PRM(pu1), PRM(pu1 + 1), T2, Nin, Ci, mid, 0);
    gnorm(T2, PRM(pu1 + 2), PRM(pu1 + 3), T2, Nin, mid, 1);
    kpconv(qp, sp, idx, T2, Nout, Nin, mid, kp, sig, PRM(pK), T3);
    gnorm(T3, PRM(pK + 1), PRM(pK + 2), T3, Nout, mid, 1);
    gemmF(T3, PRM(pu2), PRM(pu2 + 1), T4, Nout, mid, Co, 0);
    gnorm(T4, PRM(pu2 + 2), PRM(pu2 + 3), T4, Nout, Co, 0);
    const float* sc = x;
    if (strided) {
      long long tot = (long long)Nout * Ci;
      maxpool_kernel<<<(unsigned)((tot + 255) / 256), 256, 0, stream>>>(x, idx, tot, Nin, Ci, T5);
      sc = T5;
    }
    if (psc >= 0) {
      gemmF(sc, PRM(psc), PRM(psc + 1), T6, Nout, Ci, Co, 0);
      gnorm(T6, PRM(psc + 2), PRM(psc + 3), T6, Nout, Co, 0);
      sc = T6;
    }
    long long tot = (long long)Nout * Co;
    add_leaky_kernel<<<(unsigned)((tot + 255) / 256), 256, 0, stream>>>(T4, sc, outp, tot);
  };

  // --- kernel points + input features ---
  kp_gen_kernel<<<1, 1, 0, stream>>>(kpb);
  zero_kernel<<<1, 32, 0, stream>>>(meanb, 8);
  mean_reduce_kernel<<<256, 256, 0, stream>>>(pts0, CN0, meanb);
  rif_feats0_kernel<<<CN0, 64, 0, stream>>>(pts0, nb0, meanb, CN0, PRM(0), PRM(1), f0_out);
  {
    long long tot = (long long)CN0 * KNB;
    copy_int_kernel<<<(unsigned)((tot + 255) / 256), 256, 0, stream>>>(nb0, gidx_out, tot);
  }
  const float* kp1 = kpb;
  const float* kp2 = kpb + 45;
  const float* kp4 = kpb + 90;

  // --- encoder ---
  kpconv(pts0, pts0, nb0, f0_out, CN0, CN0, 64, kp1, SIGMA, PRM(2), T1);   // e1_1
  gnorm(T1, PRM(3), PRM(4), T1, CN0, 64, 1);

  resb(T1, CN0, CN0, pts0, pts0, nb0,  5,  9, 12, 16, false,  64,  32, 128, kp1, SIGMA,      F1);     // e1_2 -> feats_s1
  resb(F1, CN0, CN1, pts1, pts0, sub0, 20, 24, 27, -1, true,  128,  32, 128, kp1, SIGMA,      B1);     // e2_1
  resb(B1, CN1, CN1, pts1, pts1, nb1,  31, 35, 38, 42, false, 128,  64, 256, kp2, 2*SIGMA,    B2);     // e2_2
  resb(B2, CN1, CN1, pts1, pts1, nb1,  46, 50, 53, -1, false, 256,  64, 256, kp2, 2*SIGMA,    F2);     // e2_3 -> feats_s2
  resb(F2, CN1, CN2, pts2, pts1, sub1, 57, 61, 64, -1, true,  256,  64, 256, kp2, 2*SIGMA,    B1);     // e3_1
  resb(B1, CN2, CN2, pts2, pts2, nb2,  68, 72, 75, 79, false, 256, 128, 512, kp4, 4*SIGMA,    B2);     // e3_2
  resb(B2, CN2, CN2, pts2, pts2, nb2,  83, 87, 90, -1, false, 512, 128, 512, kp4, 4*SIGMA, f3_out);    // e3_3 -> feats_s3

  // --- decoder (reuse agg region as concat scratch; KPConv is done) ---
  float* CC  = (float*)agg;   // N1 x 768 floats (30.7 MB <= 76.8 MB)
  float* L2b = T2;            // N1 x 256
  {
    long long tot = (long long)CN1 * 768;
    gather_concat_kernel<<<(unsigned)((tot + 255) / 256), 256, 0, stream>>>(
        f3_out, up1, F2, CC, tot, 512, 256);
  }
  gemmF(CC, PRM(94), PRM(95), T3, CN1, 768, 256, 0);
  gnorm(T3, PRM(96), PRM(97), L2b, CN1, 256, 1);

  float* CC2 = (float*)agg;   // N0 x 384 floats (61.4 MB <= 76.8 MB)
  {
    long long tot = (long long)CN0 * 384;
    gather_concat_kernel<<<(unsigned)((tot + 255) / 256), 256, 0, stream>>>(
        L2b, up0, F1, CC2, tot, 256, 128);
  }
  gemmF(CC2, PRM(98), PRM(99), l1_out, CN0, 384, 256, 0);
#undef PRM
}
